// DLRM_Net_56977036149187
// MI455X (gfx1250) — matrix operand order repaired and checked by reading, not compiled, not run on hardware
//
#include <hip/hip_runtime.h>
#include <hip/hip_bf16.h>

typedef __attribute__((ext_vector_type(16))) _Float16 v16h;
typedef __attribute__((ext_vector_type(8)))  _Float16 v8h;
typedef __attribute__((ext_vector_type(8)))  float    v8f;

#define WMMA16(a, b, c) \
  __builtin_amdgcn_wmma_f32_16x16x32_f16(false, (a), false, (b), (short)0, (c), false, false)

#define ACT_RELU 0
#define ACT_SIG  1

#define DLRM_BATCH 8192
#define DLRM_NTAB  26
#define DLRM_EMBD  64
#define DLRM_NROWS 100000

__device__ __forceinline__ v16h cat8(v8h lo, v8h hi) {
  return __builtin_shufflevector(lo, hi, 0, 1, 2, 3, 4, 5, 6, 7,
                                 8, 9, 10, 11, 12, 13, 14, 15);
}

// ---------------------------------------------------------------------------
// Compile-time LUT: flat triangle index f (0..350) -> packed (M*32 + N) for
// the strict lower triangle of the 27x27 interaction matrix.
// ---------------------------------------------------------------------------
struct ZLut {
  unsigned short v[351];
};
constexpr ZLut make_zlut() {
  ZLut l{};
  int f = 0;
  for (int m = 1; m < 27; ++m)
    for (int n = 0; n < m; ++n) l.v[f++] = (unsigned short)(m * 32 + n);
  return l;
}
__device__ __constant__ ZLut ZLUT = make_zlut();

// ---------------------------------------------------------------------------
// Convert f32 -> f16 with K padding (KD >= KS, pad zeros).  KS/KD are
// compile-time: KS==KD folds to a straight copy-convert (src index == i),
// KS<KD uses a constant-divisor magic multiply instead of runtime division.
// ---------------------------------------------------------------------------
template <int KS, int KD>
__global__ __launch_bounds__(256) void cvt_pad_f16(
    const float* __restrict__ src, _Float16* __restrict__ dst, int rows) {
  const int i = blockIdx.x * 256 + threadIdx.x;
  if (i >= rows * KD) return;
  const int r = i / KD, k = i % KD;
  dst[i] = (k < KS) ? (_Float16)src[(size_t)r * KS + k] : (_Float16)0.f;
}

// ---------------------------------------------------------------------------
// Y[b, m] = act( sum_k X[b,k] * W[m,k] + bias[m] )
// X/W f16, row stride K (compile-time, multiple of 32).  M compile-time so
// store addresses fold into immediate offsets and bounds checks constant-fold.
// Each wave computes 16 batch rows x (16*NT) output cols: one A-fragment load
// feeds NT WMMAs per k-step (NT=4 -> 4x less activation re-read through L2).
// grid.x = B/128, grid.y = ceil(M / (16*NT)).
//
// Fragment layouts (wave32 gfx1250):
//  A 16x32: lanes 0-15 K{0..7,16..23}, lanes 16-31 K{8..15,24..31}
//  B 32x16: lanes 0-15 K{0..7,16..23}, lanes 16-31 K{8..15,24..31}
//  C/D:     N = lane&15, M = vgpr + ((lane&16)?8:0)
// ---------------------------------------------------------------------------
template <int K, int M, int NT, int ACT, bool OUT16>
__global__ __launch_bounds__(256) void gemm_h(
    const _Float16* __restrict__ X, const _Float16* __restrict__ W,
    const float* __restrict__ bias, void* __restrict__ Yv, int B) {
  const int lane = threadIdx.x & 31;
  const int wave = threadIdx.x >> 5;
  const int row0 = (blockIdx.x * 8 + wave) * 16;
  const int col0 = blockIdx.y * (16 * NT);
  if (row0 >= B) return;  // wave-uniform

  const int n0 = lane & 15;
  const _Float16* __restrict__ arow =
      X + (size_t)(row0 + n0) * K + ((lane & 16) ? 8 : 0);

  const _Float16* brow[NT];
#pragma unroll
  for (int nt = 0; nt < NT; ++nt) {
    const int jb = col0 + nt * 16 + n0;
    const int jsafe = ((M & 15) == 0 || jb < M) ? jb : 0;
    brow[nt] = W + (size_t)jsafe * K + ((lane & 16) ? 8 : 0);
  }

  v8f acc[NT] = {};
#pragma unroll 4
  for (int kb = 0; kb < K; kb += 32) {
    if (kb + 128 < K) __builtin_prefetch(arow + kb + 128, 0, 1);  // global_prefetch_b8
    const v16h a = cat8(*(const v8h*)(arow + kb), *(const v8h*)(arow + kb + 16));
#pragma unroll
    for (int nt = 0; nt < NT; ++nt) {
      const v16h b = cat8(*(const v8h*)(brow[nt] + kb),
                          *(const v8h*)(brow[nt] + kb + 16));
      acc[nt] = WMMA16(a, b, acc[nt]);
    }
  }

  const int mo = (lane & 16) ? 8 : 0;
#pragma unroll
  for (int nt = 0; nt < NT; ++nt) {
    const int jb = col0 + nt * 16 + n0;
    if ((M & 15) != 0 && jb >= M) continue;  // only the M=1 layer keeps this
    const float bv = bias[jb];
    if (OUT16) {
      _Float16* __restrict__ Yp = (_Float16*)Yv + (size_t)(row0 + mo) * M + jb;
#pragma unroll
      for (int v = 0; v < 8; ++v) {
        float val = acc[nt][v] + bv;
        val = (ACT == ACT_SIG) ? (1.f / (1.f + expf(-val))) : fmaxf(val, 0.f);
        Yp[(size_t)v * M] = (_Float16)val;  // immediate-offset b16 stores
      }
    } else {
      float* __restrict__ Yp = (float*)Yv + (size_t)(row0 + mo) * M + jb;
#pragma unroll
      for (int v = 0; v < 8; ++v) {
        float val = acc[nt][v] + bv;
        val = (ACT == ACT_SIG) ? (1.f / (1.f + expf(-val))) : fmaxf(val, 0.f);
        Yp[(size_t)v * M] = val;
      }
    }
  }
}

// ---------------------------------------------------------------------------
// Embedding-bag sums + pairwise interaction + assemble R = [x | tril(T T^T)]
// One sample per wave; T (27x64 padded to 32x64) staged f16 in LDS;
// Z = T T^T via 6 v_wmma (skip all-upper tile).  Z tiles are dumped to LDS
// unconditionally (ds_store_2addr_b32), then each lane copies ~11 contiguous
// flat triangle entries to R via the constant LUT -> coalesced b16 stores,
// no exec-mask juggling, no sqrt.  R is f16 [B, 416], col 415 zeroed.
// ---------------------------------------------------------------------------
__global__ __launch_bounds__(256) void emb_interact_wmma(
    const _Float16* __restrict__ xb,  // [B, 64] f16
    const int* __restrict__ lso,      // [26, B]
    const int* __restrict__ lsi,      // [26, B*4]
    const float* __restrict__ emb,    // [26, 100000, 64] f32
    _Float16* __restrict__ R) {       // [B, 416] f16
  __shared__ _Float16 Tl[8][32][DLRM_EMBD];  // 32 KB
  __shared__ float    Zl[8][32][32];         // 32 KB
  const int lane = threadIdx.x & 31;
  const int wave = threadIdx.x >> 5;
  const int b = blockIdx.x * 8 + wave;
  _Float16* __restrict__ Tp = &Tl[wave][0][0];
  const int c0 = 2 * lane;

  // Row 0 of T = bottom-MLP activation; also first 64 cols of R.
  const _Float16 x0 = xb[(size_t)b * DLRM_EMBD + c0];
  const _Float16 x1 = xb[(size_t)b * DLRM_EMBD + c0 + 1];
  Tp[c0] = x0;
  Tp[c0 + 1] = x1;
  R[(size_t)b * 416 + c0] = x0;
  R[(size_t)b * 416 + c0 + 1] = x1;
  if (lane == 0) R[(size_t)b * 416 + 415] = (_Float16)0.f;  // zero pad col

  // Rows 1..26: embedding-bag sums. One float2 (8B) per lane per pooled row
  // -> the wave pulls the full 256B embedding row. HBM-bound part (~218 MB).
  for (int t = 0; t < DLRM_NTAB; ++t) {
    const int* __restrict__ off = lso + (size_t)t * DLRM_BATCH;
    const int s = off[b];
    const int e = (b + 1 < DLRM_BATCH) ? off[b + 1] : DLRM_BATCH * 4;
    const int* __restrict__ idx = lsi + (size_t)t * DLRM_BATCH * 4;
    const float2* __restrict__ tab2 =
        (const float2*)(emb + (size_t)t * DLRM_NROWS * DLRM_EMBD);
    float s0 = 0.f, s1 = 0.f;
    for (int p = s; p < e; ++p) {
      const float2 v = tab2[(size_t)idx[p] * 32 + lane];
      s0 += v.x;
      s1 += v.y;
    }
    Tp[(1 + t) * DLRM_EMBD + c0] = (_Float16)s0;
    Tp[(1 + t) * DLRM_EMBD + c0 + 1] = (_Float16)s1;
  }
#pragma unroll
  for (int r = 27; r < 32; ++r) {  // zero-pad rows 27..31
    Tp[r * DLRM_EMBD + c0] = (_Float16)0.f;
    Tp[r * DLRM_EMBD + c0 + 1] = (_Float16)0.f;
  }
  __syncthreads();

  // WMMA fragments of T from LDS (ds_load_b128, no guards).
  v16h afr[2][2], bfr[2][2];
#pragma unroll
  for (int mt = 0; mt < 2; ++mt)
#pragma unroll
    for (int ks = 0; ks < 2; ++ks) {
      const _Float16* ar = Tp + (mt * 16 + (lane & 15)) * DLRM_EMBD +
                           ks * 32 + ((lane & 16) ? 8 : 0);
      afr[mt][ks] = cat8(*(const v8h*)ar, *(const v8h*)(ar + 16));
    }
#pragma unroll
  for (int nt = 0; nt < 2; ++nt)
#pragma unroll
    for (int ks = 0; ks < 2; ++ks) {
      const _Float16* br = Tp + (nt * 16 + (lane & 15)) * DLRM_EMBD +
                           ks * 32 + ((lane & 16) ? 8 : 0);
      bfr[nt][ks] = cat8(*(const v8h*)br, *(const v8h*)(br + 16));
    }

  // Z = T T^T; dump the 3 needed 16x16 tiles to LDS (unconditional stores).
  const int mo = (lane & 16) ? 8 : 0;
  const int n0 = (lane & 15);
#pragma unroll
  for (int mt = 0; mt < 2; ++mt)
#pragma unroll
    for (int nt = 0; nt < 2; ++nt) {
      if (mt == 0 && nt == 1) continue;  // entirely upper triangle
      v8f acc = {};
      acc = WMMA16(afr[mt][0], bfr[nt][0], acc);
      acc = WMMA16(afr[mt][1], bfr[nt][1], acc);
      float* __restrict__ zrow = &Zl[wave][mt * 16 + mo][nt * 16 + n0];
#pragma unroll
      for (int v = 0; v < 8; ++v) zrow[(size_t)v * 32] = acc[v];
    }

  // Compact strict lower triangle (351 entries) -> R cols 64..414.
  // Same-wave LDS ops are in-order, so no barrier needed (wave-private tile).
  const float* __restrict__ Zw = &Zl[wave][0][0];
  for (int f = lane; f < 351; f += 32) {
    R[(size_t)b * 416 + 64 + f] = (_Float16)Zw[ZLUT.v[f]];
  }
}

// ---------------------------------------------------------------------------
// Launch
// ---------------------------------------------------------------------------
extern "C" void kernel_launch(void* const* d_in, const int* in_sizes, int n_in,
                              void* d_out, int out_size, void* d_ws, size_t ws_size,
                              hipStream_t stream) {
  (void)in_sizes; (void)n_in; (void)out_size; (void)ws_size;

  const float* dense_x = (const float*)d_in[0];   // [8192, 13]
  const int*   lS_o    = (const int*)  d_in[1];   // [26, 8192]
  const int*   lS_i    = (const int*)  d_in[2];   // [26, 32768]
  const float* emb_w   = (const float*)d_in[3];   // [26, 100000, 64]
  const float* wb1     = (const float*)d_in[4];   // [512, 13]
  const float* wb2     = (const float*)d_in[5];   // [256, 512]
  const float* wb3     = (const float*)d_in[6];   // [64, 256]
  const float* bb1     = (const float*)d_in[7];
  const float* bb2     = (const float*)d_in[8];
  const float* bb3     = (const float*)d_in[9];
  const float* wt1     = (const float*)d_in[10];  // [512, 415]
  const float* wt2     = (const float*)d_in[11];  // [256, 512]
  const float* wt3     = (const float*)d_in[12];  // [1, 256]
  const float* bt1     = (const float*)d_in[13];
  const float* bt2     = (const float*)d_in[14];
  const float* bt3     = (const float*)d_in[15];

  // Workspace partition (256B aligned, all f16)
  char* base = (char*)d_ws;
  auto alloc16 = [&](size_t nelem) {
    _Float16* p = (_Float16*)base;
    base += ((nelem * sizeof(_Float16)) + 255) & ~(size_t)255;
    return p;
  };
  _Float16* xpad = alloc16((size_t)DLRM_BATCH * 32);    // padded dense_x
  _Float16* y1   = alloc16((size_t)DLRM_BATCH * 512);
  _Float16* y2   = alloc16((size_t)DLRM_BATCH * 256);
  _Float16* xb   = alloc16((size_t)DLRM_BATCH * 64);
  _Float16* R    = alloc16((size_t)DLRM_BATCH * 416);
  _Float16* t1   = alloc16((size_t)DLRM_BATCH * 512);
  _Float16* t2   = alloc16((size_t)DLRM_BATCH * 256);
  _Float16* hw1  = alloc16((size_t)512 * 32);
  _Float16* hw2  = alloc16((size_t)256 * 512);
  _Float16* hw3  = alloc16((size_t)64 * 256);
  _Float16* hwt1 = alloc16((size_t)512 * 416);
  _Float16* hwt2 = alloc16((size_t)256 * 512);
  _Float16* hwt3 = alloc16((size_t)1 * 256);
  float* out = (float*)d_out;  // [8192, 1]

  const dim3 blk(256);
  auto g1 = [](size_t n) { return dim3((unsigned)((n + 255) / 256)); };

  // f32 -> f16 (+K pad) conversions (KS/KD compile-time: no runtime division)
  cvt_pad_f16<13, 32>  <<<g1((size_t)DLRM_BATCH * 32), blk, 0, stream>>>(dense_x, xpad, DLRM_BATCH);
  cvt_pad_f16<13, 32>  <<<g1((size_t)512 * 32),  blk, 0, stream>>>(wb1, hw1, 512);
  cvt_pad_f16<512, 512><<<g1((size_t)256 * 512), blk, 0, stream>>>(wb2, hw2, 256);
  cvt_pad_f16<256, 256><<<g1((size_t)64 * 256),  blk, 0, stream>>>(wb3, hw3, 64);
  cvt_pad_f16<415, 416><<<g1((size_t)512 * 416), blk, 0, stream>>>(wt1, hwt1, 512);
  cvt_pad_f16<512, 512><<<g1((size_t)256 * 512), blk, 0, stream>>>(wt2, hwt2, 256);
  cvt_pad_f16<256, 256><<<g1((size_t)1 * 256),   blk, 0, stream>>>(wt3, hwt3, 1);

  const int gx = DLRM_BATCH / 128;  // 64 blocks along batch

  // Bottom MLP (all ReLU)
  gemm_h<32,  512, 4, ACT_RELU, true><<<dim3(gx, 8), blk, 0, stream>>>(xpad, hw1, bb1, y1, DLRM_BATCH);
  gemm_h<512, 256, 4, ACT_RELU, true><<<dim3(gx, 4), blk, 0, stream>>>(y1,   hw2, bb2, y2, DLRM_BATCH);
  gemm_h<256, 64,  4, ACT_RELU, true><<<dim3(gx, 1), blk, 0, stream>>>(y2,   hw3, bb3, xb, DLRM_BATCH);

  // Embedding bags + interaction
  emb_interact_wmma<<<dim3(DLRM_BATCH / 8), blk, 0, stream>>>(xb, lS_o, lS_i, emb_w, R);

  // Top MLP (ReLU, ReLU, Sigmoid->f32 out)
  gemm_h<416, 512, 4, ACT_RELU, true ><<<dim3(gx, 8), blk, 0, stream>>>(R,  hwt1, bt1, t1,  DLRM_BATCH);
  gemm_h<512, 256, 4, ACT_RELU, true ><<<dim3(gx, 4), blk, 0, stream>>>(t1, hwt2, bt2, t2,  DLRM_BATCH);
  gemm_h<256, 1,   1, ACT_SIG,  false><<<dim3(gx, 1), blk, 0, stream>>>(t2, hwt3, bt3, out, DLRM_BATCH);
}
